// LSTM_17257178595688
// MI455X (gfx1250) — compile-verified, matrix operand order checked by reading, not run on hardware
//
#include <hip/hip_runtime.h>
#include <hip/hip_bf16.h>

// ---------------------------------------------------------------------------
// LSTM on gfx1250 (MI455X), wave32 + WMMA + LDS double-buffered staging.
//
//  * All GEMM work on v_wmma_f32_16x16x32_f16 (fp16 operands, fp32 accum).
//  * Operands pre-packed into the per-lane WMMA fragment layout (ISA 7.12.2)
//    so hot loops are global_b128/ds_b128 loads + wmma only.
//  * Gate kernel: one block = one hidden-column tile (ct), 4 waves = 4 batch
//    tiles (mt). The 4 gate B-fragments per K-step are shared by all waves,
//    staged through an 8 KB LDS double buffer; the next stage's global load
//    is issued before the current stage's wmmas (software pipeline).
//  * K loop split into x-phase / h-phase via an inlined lambda taking the
//    phase's A pointer directly -> address-space inference keeps A loads on
//    the GLOBAL path (flat loads would also burn DScnt + LDS bandwidth).
//  * Output projection batched into one big GEMM at the end; its waves are
//    grouped so a block shares one W_out column tile (WGP$ reuse).
//
// Workspace layout (bytes, total ~154 MB):
//   [0)            Wg16   : packed W_g fragments   (4096x2048 f16) 16,777,216
//   [16777216)     Wout16 : packed W_out fragments (1024x1024 f16)  2,097,152
//   [18874368)     Ax     : packed x fragments, 512 slots          67,108,864
//   [85983232)     Ah     : packed h fragments, 513 slots          67,239,936
//   [153223168)    Cbuf   : cell state fp32 (64x1024)                 262,144
//   [153485312)    Hrow   : h_t row-major fp32 (64x1024)              262,144
// ---------------------------------------------------------------------------

typedef _Float16 half_t;
typedef __attribute__((ext_vector_type(16))) _Float16 v16h;
typedef __attribute__((ext_vector_type(8)))  float    v8f;

#define NB   64        // batch
#define SL   512       // sequence length
#define IS   1024      // input size
#define HS   1024      // hidden size
#define SLOT 65536     // halfs per packed 64x1024 A slot (128 frags * 512)

static __device__ __forceinline__ float fast_sigmoid(float x) {
    return 1.0f / (1.0f + __expf(-x));
}
static __device__ __forceinline__ float fast_tanh(float x) {
    float t = __expf(2.0f * x);
    return (t - 1.0f) / (t + 1.0f);
}

// Fragment element mapping (ISA 7.12.2, 16-bit A 16x32 / B 32x16):
//   lane l: p = l & 15 (row M for A / col N for B)
//   half h: k = (h < 8 ? h : h + 8) + (l >= 16 ? 8 : 0)
// Fragments stored lane-major: frag*512 + lane*16 halfs (32 B/lane).

__global__ void __launch_bounds__(256)
pack_tiles_f32_to_f16(const float* __restrict__ src, half_t* __restrict__ dst,
                      size_t ld, int ptiles, int ktiles) {
    int tid  = blockIdx.x * blockDim.x + threadIdx.x;
    int lane = tid & 31;
    int frag = tid >> 5;
    if (frag >= ptiles * ktiles) return;
    int kt = frag % ktiles;
    int pt = frag / ktiles;
    int p    = lane & 15;
    int koff = (lane >= 16) ? 8 : 0;
    const float* s = src + (size_t)(pt * 16 + p) * ld + (size_t)kt * 32 + koff;
    v16h v;
#pragma unroll
    for (int h = 0; h < 16; ++h) {
        int k = (h < 8) ? h : (h + 8);
        v[h] = (half_t)s[k];
    }
    *(v16h*)(dst + (size_t)frag * 512 + lane * 16) = v;
}

// Pack x (N, L, IS) into per-timestep A fragments: slot t has frags (mt*32+kt).
__global__ void __launch_bounds__(256)
pack_x_kernel(const float* __restrict__ x, half_t* __restrict__ dst) {
    int t    = blockIdx.y;
    int tid  = blockIdx.x * blockDim.x + threadIdx.x;   // 4096 per t
    int lane = tid & 31;
    int frag = tid >> 5;                                // 0..127
    if (frag >= 128) return;
    int kt = frag & 31;
    int mt = frag >> 5;
    int p    = lane & 15;
    int koff = (lane >= 16) ? 8 : 0;
    const float* s = x + ((size_t)(mt * 16 + p) * SL + t) * IS + (size_t)kt * 32 + koff;
    v16h v;
#pragma unroll
    for (int h = 0; h < 16; ++h) {
        int k = (h < 8) ? h : (h + 8);
        v[h] = (half_t)s[k];
    }
    *(v16h*)(dst + (size_t)t * SLOT + (size_t)frag * 512 + lane * 16) = v;
}

// One recurrent step. Grid: 64 blocks (one per ct), 128 threads (4 waves).
// Wave w = batch tile mt; all waves share ct => share the 4 gate B-frags,
// which are double-buffered through LDS.
__global__ void __launch_bounds__(128)
lstm_gate_step(const half_t* __restrict__ Ax,   // slot t of packed x frags
               const half_t* __restrict__ Ah,   // slot t of packed h frags (= h_{t-1})
               const half_t* __restrict__ Wg,   // frag (nt*64 + kt), nt = g*64+ct
               const float* __restrict__ bf, const float* __restrict__ bi,
               const float* __restrict__ bc, const float* __restrict__ bo,
               float* __restrict__ Cbuf, float* __restrict__ Hrow) {
    __shared__ __align__(32) half_t Bsh[2][4][512];   // [buf][gate][frag] 8 KB

    const int tid   = threadIdx.x;       // 0..127
    const int lane  = tid & 31;
    const int mt    = tid >> 5;          // wave id = batch tile
    const int ct    = blockIdx.x;        // hidden column tile
    const int gld   = tid >> 5;          // staging role: gate index
    const int chunk = tid & 31;          // staging role: 32 B chunk
    const int col   = ct * 16 + (lane & 15);

    // Staging source base for (gate gld, ct): frag index ((gld*64+ct)*64 + kt).
    const half_t* WgStage = Wg + ((size_t)((gld * 64 + ct) * 64) * 512) + chunk * 16;

    // Fold biases into accumulator init (C layout: lane -> column).
    float vbf = bf[col], vbi = bi[col], vbc = bc[col], vbo = bo[col];
    v8f aF, aI, aC, aO;
#pragma unroll
    for (int r = 0; r < 8; ++r) { aF[r] = vbf; aI[r] = vbi; aC[r] = vbc; aO[r] = vbo; }

    // Prologue: stage kt = 0 into buffer 0.
    {
        v16h v = *(const v16h*)(WgStage);                 // kt = 0
        *(v16h*)(&Bsh[0][gld][0] + chunk * 16) = v;
    }
    __syncthreads();

    // One K-phase (32 steps) over a directly-passed A base pointer; inlined at
    // both call sites so the A loads stay rooted at the kernel args (GLOBAL).
    auto run_phase = [&](const half_t* __restrict__ Ab, int base) {
#pragma unroll 2
        for (int kk = 0; kk < 32; ++kk) {
            const int kt  = base + kk;
            const int ktn = kt + 1;
            const bool hn = ktn < 64;

            // Issue next stage's global load first (pipelined past the wmmas).
            v16h nxt;
            if (hn) nxt = *(const v16h*)(WgStage + (size_t)ktn * 512);

            // Per-wave A fragment (x or h part), straight-line addressing.
            v16h a = *(const v16h*)(Ab + (size_t)(mt * 32 + kk) * 512 + lane * 16);

            // Shared B fragments from LDS.
            const half_t* bb = &Bsh[kt & 1][0][0];
            v16h b0 = *(const v16h*)(bb + 0 * 512 + lane * 16);
            v16h b1 = *(const v16h*)(bb + 1 * 512 + lane * 16);
            v16h b2 = *(const v16h*)(bb + 2 * 512 + lane * 16);
            v16h b3 = *(const v16h*)(bb + 3 * 512 + lane * 16);

            aF = __builtin_amdgcn_wmma_f32_16x16x32_f16(false, a, false, b0, (short)0, aF, false, false);
            aI = __builtin_amdgcn_wmma_f32_16x16x32_f16(false, a, false, b1, (short)0, aI, false, false);
            aC = __builtin_amdgcn_wmma_f32_16x16x32_f16(false, a, false, b2, (short)0, aC, false, false);
            aO = __builtin_amdgcn_wmma_f32_16x16x32_f16(false, a, false, b3, (short)0, aO, false, false);

            // Park the prefetched stage in the other LDS buffer.
            if (hn) *(v16h*)(&Bsh[ktn & 1][gld][0] + chunk * 16) = nxt;
            __syncthreads();
        }
    };
    run_phase(Ax, 0);    // K = 0..1023   : x_t contribution
    run_phase(Ah, 32);   // K = 1024..2047: h_{t-1} contribution

    // Pointwise cell update on the C-layout registers:
    // VGPR r, lane l -> row m = r + (l>=16 ? 8 : 0), col n = l & 15.
    int hi = (lane >= 16) ? 8 : 0;
#pragma unroll
    for (int r = 0; r < 8; ++r) {
        int nb = mt * 16 + r + hi;                 // batch index
        size_t ix = (size_t)nb * HS + col;
        float f = fast_sigmoid(aF[r]);
        float i = fast_sigmoid(aI[r]);
        float g = fast_tanh(aC[r]);
        float o = fast_sigmoid(aO[r]);
        float c = f * Cbuf[ix] + i * g;
        Cbuf[ix] = c;
        Hrow[ix] = o * fast_tanh(c);
    }
}

// Batched output projection: Y[t,n,:] = h_t[n,:] @ W_out^T + b_out for all t.
// 131072 waves; waves are grouped so one block (8 waves) shares a single W_out
// column tile ct -> B fragments hit in WGP$ after the first wave touches them.
__global__ void __launch_bounds__(256)
out_gemm(const half_t* __restrict__ Ah,     // base of slot for h_1 (t=0 output)
         const half_t* __restrict__ Wo,     // frag (ct*32 + kt)
         const float* __restrict__ bout, float* __restrict__ y) {
    size_t tid  = (size_t)blockIdx.x * blockDim.x + threadIdx.x;
    int    lane = (int)(tid & 31);
    size_t wid  = tid >> 5;                 // 0..131071
    int ct  = (int)(wid >> 11);             // 2048 waves per ct, contiguous
    int rem = (int)(wid & 2047);
    int mt  = rem & 3;
    int t   = rem >> 2;                     // 0..511
    const half_t* A = Ah + (size_t)t * SLOT;
    int col = ct * 16 + (lane & 15);

    float vb = bout[col];
    v8f acc;
#pragma unroll
    for (int r = 0; r < 8; ++r) acc[r] = vb;

#pragma unroll 4
    for (int kt = 0; kt < 32; ++kt) {
        v16h a = *(const v16h*)(A  + (size_t)(mt * 32 + kt) * 512 + lane * 16);
        v16h b = *(const v16h*)(Wo + (size_t)(ct * 32 + kt) * 512 + lane * 16);
        acc = __builtin_amdgcn_wmma_f32_16x16x32_f16(false, a, false, b, (short)0, acc, false, false);
    }

    int hi = (lane >= 16) ? 8 : 0;
#pragma unroll
    for (int r = 0; r < 8; ++r) {
        int nb = mt * 16 + r + hi;          // batch index
        y[((size_t)nb * SL + t) * HS + col] = acc[r];
    }
}

// Final c and h appended after y in d_out (reference returns (y, c, h)).
__global__ void tail_copy(const float* __restrict__ C, const float* __restrict__ H,
                          float* __restrict__ out) {
    int i = blockIdx.x * blockDim.x + threadIdx.x;
    if (i < NB * HS) {
        out[i]           = C[i];
        out[NB * HS + i] = H[i];
    }
}

extern "C" void kernel_launch(void* const* d_in, const int* in_sizes, int n_in,
                              void* d_out, int out_size, void* d_ws, size_t ws_size,
                              hipStream_t stream) {
    const float* x     = (const float*)d_in[0];
    const float* c0    = (const float*)d_in[1];
    const float* h0    = (const float*)d_in[2];
    const float* W_f   = (const float*)d_in[3];
    const float* b_f   = (const float*)d_in[4];
    const float* W_i   = (const float*)d_in[5];
    const float* b_i   = (const float*)d_in[6];
    const float* W_c   = (const float*)d_in[7];
    const float* b_c   = (const float*)d_in[8];
    const float* W_o   = (const float*)d_in[9];
    const float* b_o   = (const float*)d_in[10];
    const float* W_out = (const float*)d_in[11];
    const float* b_out = (const float*)d_in[12];
    float* y = (float*)d_out;

    char*   ws     = (char*)d_ws;
    half_t* Wg16   = (half_t*)(ws);
    half_t* Wout16 = (half_t*)(ws + 16777216);
    half_t* Ax     = (half_t*)(ws + 18874368);
    half_t* Ah     = (half_t*)(ws + 85983232);   // 513 slots; slot 0 = h0
    float*  Cbuf   = (float*)(ws + 153223168);
    float*  Hrow   = (float*)(ws + 153485312);

    // --- Pack weights into WMMA fragment layout (recomputed every call) ---
    const size_t gateFrags = (size_t)64 * 64;            // per gate section
    pack_tiles_f32_to_f16<<<512, 256, 0, stream>>>(W_f, Wg16 + 0 * gateFrags * 512, 2048, 64, 64);
    pack_tiles_f32_to_f16<<<512, 256, 0, stream>>>(W_i, Wg16 + 1 * gateFrags * 512, 2048, 64, 64);
    pack_tiles_f32_to_f16<<<512, 256, 0, stream>>>(W_c, Wg16 + 2 * gateFrags * 512, 2048, 64, 64);
    pack_tiles_f32_to_f16<<<512, 256, 0, stream>>>(W_o, Wg16 + 3 * gateFrags * 512, 2048, 64, 64);
    pack_tiles_f32_to_f16<<<256, 256, 0, stream>>>(W_out, Wout16, 1024, 64, 32);

    // --- Pack all x timesteps (parallel, off the critical chain) ---
    pack_x_kernel<<<dim3(16, SL), 256, 0, stream>>>(x, Ax);

    // --- Init recurrent state ---
    hipMemcpyAsync(Cbuf, c0, (size_t)NB * HS * sizeof(float),
                   hipMemcpyDeviceToDevice, stream);
    pack_tiles_f32_to_f16<<<16, 256, 0, stream>>>(h0, Ah, 1024, 4, 32);  // slot 0

    // --- Recurrent loop: only the gate GEMM + cell update is sequential ---
    for (int t = 0; t < SL; ++t) {
        lstm_gate_step<<<64, 128, 0, stream>>>(
            Ax + (size_t)t * SLOT, Ah + (size_t)t * SLOT, Wg16,
            b_f, b_i, b_c, b_o, Cbuf, Hrow);
        // Pack h_t into slot t+1 (consumed by step t+1 and by out_gemm).
        pack_tiles_f32_to_f16<<<16, 256, 0, stream>>>(
            Hrow, Ah + (size_t)(t + 1) * SLOT, 1024, 4, 32);
    }

    // --- Batched output projection over all 512 timesteps ---
    out_gemm<<<16384, 256, 0, stream>>>(Ah + SLOT, Wout16, b_out, y);

    // --- Append final c, h ---
    tail_copy<<<256, 256, 0, stream>>>(Cbuf, Hrow, y + (size_t)NB * SL * HS);
}
